// GraniteMoeHybridMambaDecoderLayer_40724879901178
// MI455X (gfx1250) — matrix operand order repaired and checked by reading, not compiled, not run on hardware
//
#include <hip/hip_runtime.h>
#include <hip/hip_bf16.h>
#include <math.h>

// ---------------------------------------------------------------------------
// MI455X (gfx1250) fused Mamba2+MoE decoder layer.
// Compute-bound (~1.2 TFLOP vs ~0.6GB traffic) -> all large GEMMs go through
// v_wmma_f32_16x16x32_bf16 (wave32 WMMA), f32 accumulate, bf16 staged in LDS.
// Weights are pre-converted to bf16 once per launch (halves weight-side L2
// traffic, removes all conversion VALU from the GEMM K-loop). B tiles are
// stored TRANSPOSED in LDS so every WMMA operand fragment is two contiguous
// 16B runs -> 2x ds_load_b128 per fragment, no repacking.
// ---------------------------------------------------------------------------

typedef __bf16         bf16x16 __attribute__((ext_vector_type(16)));
typedef __bf16         bf16x8  __attribute__((ext_vector_type(8)));
typedef float          f32x8   __attribute__((ext_vector_type(8)));
typedef float          f32x4   __attribute__((ext_vector_type(4)));
typedef unsigned short u16x8   __attribute__((ext_vector_type(8)));
typedef unsigned short u16x4   __attribute__((ext_vector_type(4)));

#define DMODEL   2048
#define DINNER   4096
#define NH       64
#define PDIM     64
#define NSTATE   128
#define KCONV    4
#define CHUNK    256
#define NEXP     8
#define IEXP     1024
#define ISH      4096
#define CONVDIM  (DINNER + 2*NSTATE)        // 4352
#define PROJW    (2*DINNER + 2*NSTATE + NH) // 8512
#define EPSV     1e-5f
#define RESM     0.22f

static __device__ __forceinline__ unsigned short f2bf(float f) {
  unsigned u = __builtin_bit_cast(unsigned, f);
  unsigned r = u + 0x7FFFu + ((u >> 16) & 1u);     // round-to-nearest-even
  return (unsigned short)(r >> 16);
}
static __device__ __forceinline__ float bf2f(unsigned short h) {
  unsigned u = ((unsigned)h) << 16;
  return __builtin_bit_cast(float, u);
}
static __device__ __forceinline__ float siluf(float x) {
  return x / (1.0f + __expf(-x));
}

static __device__ __forceinline__ f32x8 wmma_bf16(bf16x16 a, bf16x16 b, f32x8 c) {
  // (neg_a, A, neg_b, B, c_mod, C, reuse_a, reuse_b) -> v_wmma_f32_16x16x32_bf16
  return __builtin_amdgcn_wmma_f32_16x16x32_bf16(false, a, false, b, (short)0, c, false, false);
}

// ---- LDS fragment loaders (ISA 7.12.2 VGPR layouts, wave32) ---------------
// A tile in LDS: [m][k], row stride 40 u16 (80B: rows stay 16B aligned).
// A 16x32 bf16: lane l<16 -> row M=l, K = {0..7, 16..23}; lane>=16 -> K={8..15,24..31}
static __device__ __forceinline__ bf16x16 frag_a(const unsigned short* base, int ldk, int lane) {
  const __bf16* p = (const __bf16*)(base + (lane & 15) * ldk + ((lane < 16) ? 0 : 8));
  bf16x8 lo = *(const bf16x8*)(p);
  bf16x8 hi = *(const bf16x8*)(p + 16);
  return __builtin_shufflevector(lo, hi, 0,1,2,3,4,5,6,7,8,9,10,11,12,13,14,15);
}
// B tile stored TRANSPOSED in LDS: Bt[n][k], row stride 40 u16.
// B 32x16 bf16: lanes 0-15 hold K={0..7,16..23}, lanes 16-31 hold K={8..15,24..31}; N = lane&15
static __device__ __forceinline__ bf16x16 frag_bT(const unsigned short* base, int lane) {
  const __bf16* p = (const __bf16*)(base + (lane & 15) * 40 + ((lane < 16) ? 0 : 8));
  bf16x8 lo = *(const bf16x8*)(p);
  bf16x8 hi = *(const bf16x8*)(p + 16);
  return __builtin_shufflevector(lo, hi, 0,1,2,3,4,5,6,7,8,9,10,11,12,13,14,15);
}

// ---------------------------------------------------------------------------
// Bulk f32 -> bf16 weight conversion (once per launch; ~26us of HBM traffic).
// ---------------------------------------------------------------------------
__global__ __launch_bounds__(256)
void cvt_bf16(const float* __restrict__ in, unsigned short* __restrict__ out, long long n)
{
  long long i = ((long long)blockIdx.x * 256 + threadIdx.x) * 8;
  if (i + 8 > n) return;
  f32x4 a = *(const f32x4*)(in + i);
  f32x4 b = *(const f32x4*)(in + i + 4);
  u16x8 o;
#pragma unroll
  for (int j = 0; j < 4; ++j) { o[j] = f2bf(a[j]); o[4 + j] = f2bf(b[j]); }
  *(u16x8*)(out + i) = o;
}

// ---------------------------------------------------------------------------
// Generic tiled WMMA GEMM: C[M,N](f32) = A[M,K] * B[K,N]
// A: bf16 or f32 (converted on stage); B: bf16 (copy) or f32 (converted),
// optional transpose (B stored [N,K] row-major).
// Block 128x128x32, 8 waves (4x2), wave 32x64. blockIdx.z batches with
// element strides sA/sB/sC.  N must be a multiple of 8.
// ---------------------------------------------------------------------------
template<bool A_BF16, bool B_BF16, bool B_TRANS>
__global__ __launch_bounds__(256)
void gemm_wmma(const void* __restrict__ Aptr, const void* __restrict__ Bptr,
               float* __restrict__ Cptr, int M, int N, int K,
               int lda, int ldb, int ldc,
               long long sA, long long sB, long long sC)
{
  __shared__ __align__(16) unsigned short As[128 * 40];   // [m][k]
  __shared__ __align__(16) unsigned short Bs[128 * 40];   // [n][k] (transposed)

  const int tid  = threadIdx.x;
  const int lane = tid & 31;
  const int wave = tid >> 5;
  const int wm = wave >> 1;                 // 0..3
  const int wn = wave & 1;                  // 0..1
  const int bM = blockIdx.y * 128;
  const int bN = blockIdx.x * 128;

  float* Cb = Cptr + (long long)blockIdx.z * sC;
  const unsigned short* Ab16 = (const unsigned short*)Aptr + (A_BF16 ? (long long)blockIdx.z * sA : 0);
  const float*          Ab32 = (const float*)Aptr          + (A_BF16 ? 0 : (long long)blockIdx.z * sA);
  const unsigned short* Bb16 = (const unsigned short*)Bptr + (B_BF16 ? (long long)blockIdx.z * sB : 0);
  const float*          Bb32 = (const float*)Bptr          + (B_BF16 ? 0 : (long long)blockIdx.z * sB);

  f32x8 acc[2][4];
#pragma unroll
  for (int i = 0; i < 2; ++i)
#pragma unroll
    for (int j = 0; j < 4; ++j)
#pragma unroll
      for (int r = 0; r < 8; ++r) acc[i][j][r] = 0.0f;

  for (int k0 = 0; k0 < K; k0 += 32) {
    __syncthreads();
    // ---- stage A tile: 128x32 ----
    if (A_BF16) {
#pragma unroll
      for (int g = 0; g < 2; ++g) {
        int gi = tid + 256 * g;                  // 512 groups of 8
        int row = gi >> 2, ko = (gi & 3) * 8;
        int gm = bM + row;
        u16x8 v = {0,0,0,0,0,0,0,0};
        if (gm < M) v = *(const u16x8*)(Ab16 + (size_t)gm * lda + k0 + ko);
        *(u16x8*)(&As[row * 40 + ko]) = v;
      }
    } else {
#pragma unroll
      for (int g = 0; g < 4; ++g) {
        int gi = tid + 256 * g;                  // 1024 groups of 4
        int row = gi >> 3, ko = (gi & 7) * 4;
        int gm = bM + row;
        f32x4 v = {0.f,0.f,0.f,0.f};
        if (gm < M) v = *(const f32x4*)(Ab32 + (size_t)gm * lda + k0 + ko);
        u16x4 h;
#pragma unroll
        for (int i = 0; i < 4; ++i) h[i] = f2bf(v[i]);
        *(u16x4*)(&As[row * 40 + ko]) = h;
      }
    }
    // ---- stage B tile into transposed LDS Bt[n][k] ----
    if (B_BF16) {
      if (B_TRANS) {                             // source bf16 B[N][K]
#pragma unroll
        for (int g = 0; g < 2; ++g) {
          int gi = tid + 256 * g;                // n = gi>>2, k8 = (gi&3)*8
          int n = gi >> 2, k8 = (gi & 3) * 8;
          int gn = bN + n;
          u16x8 v = {0,0,0,0,0,0,0,0};
          if (gn < N) v = *(const u16x8*)(Bb16 + (size_t)gn * ldb + k0 + k8);
          *(u16x8*)(&Bs[n * 40 + k8]) = v;
        }
      } else {                                   // source bf16 B[K][N]
#pragma unroll
        for (int g = 0; g < 2; ++g) {
          int gi = tid + 256 * g;                // k = gi>>4, n8 = (gi&15)*8
          int k = gi >> 4, n8 = (gi & 15) * 8;
          int gn = bN + n8;
          u16x8 v = {0,0,0,0,0,0,0,0};
          if (gn < N) v = *(const u16x8*)(Bb16 + (size_t)(k0 + k) * ldb + gn);
#pragma unroll
          for (int i = 0; i < 8; ++i) Bs[(n8 + i) * 40 + k] = v[i];
        }
      }
    } else {
      if (B_TRANS) {                             // source f32 B[N][K]
#pragma unroll
        for (int g = 0; g < 4; ++g) {
          int gi = tid + 256 * g;                // n = gi>>3, k4 = (gi&7)*4
          int n = gi >> 3, k4 = (gi & 7) * 4;
          int gn = bN + n;
          f32x4 v = {0.f,0.f,0.f,0.f};
          if (gn < N) v = *(const f32x4*)(Bb32 + (size_t)gn * ldb + k0 + k4);
          u16x4 h;
#pragma unroll
          for (int i = 0; i < 4; ++i) h[i] = f2bf(v[i]);
          *(u16x4*)(&Bs[n * 40 + k4]) = h;
        }
      } else {                                   // source f32 B[K][N]
#pragma unroll
        for (int g = 0; g < 4; ++g) {
          int gi = tid + 256 * g;                // k = gi>>5, n4 = (gi&31)*4
          int k = gi >> 5, n4 = (gi & 31) * 4;
          int gn = bN + n4;
          f32x4 v = {0.f,0.f,0.f,0.f};
          if (gn < N) v = *(const f32x4*)(Bb32 + (size_t)(k0 + k) * ldb + gn);
#pragma unroll
          for (int i = 0; i < 4; ++i) Bs[(n4 + i) * 40 + k] = f2bf(v[i]);
        }
      }
    }
    // prefetch next weight tile into L2 (global_prefetch_b8)
    if (k0 + 32 < K) {
      const void* nxt;
      if (B_TRANS) nxt = B_BF16 ? (const void*)(Bb16 + (size_t)(bN + (tid & 127)) * ldb + k0 + 32)
                                : (const void*)(Bb32 + (size_t)(bN + (tid & 127)) * ldb + k0 + 32);
      else         nxt = B_BF16 ? (const void*)(Bb16 + (size_t)(k0 + 32 + (tid >> 7)) * ldb + bN + (tid & 127))
                                : (const void*)(Bb32 + (size_t)(k0 + 32 + (tid >> 7)) * ldb + bN + (tid & 127));
      __builtin_prefetch(nxt, 0, 1);
    }
    __syncthreads();

    bf16x16 af[2], bfv[4];
#pragma unroll
    for (int i = 0; i < 2; ++i) af[i] = frag_a(As + (wm * 32 + i * 16) * 40, 40, lane);
#pragma unroll
    for (int j = 0; j < 4; ++j) bfv[j] = frag_bT(Bs + (wn * 64 + j * 16) * 40, lane);
#pragma unroll
    for (int i = 0; i < 2; ++i)
#pragma unroll
      for (int j = 0; j < 4; ++j)
        acc[i][j] = wmma_bf16(af[i], bfv[j], acc[i][j]);
  }

  // ---- store: VGPR r -> M = base + r + 8*(lane>=16); N = base + (lane&15)
#pragma unroll
  for (int i = 0; i < 2; ++i) {
#pragma unroll
    for (int j = 0; j < 4; ++j) {
      int row0 = bM + wm * 32 + i * 16 + ((lane < 16) ? 0 : 8);
      int col  = bN + wn * 64 + j * 16 + (lane & 15);
      if (col < N) {
#pragma unroll
        for (int r = 0; r < 8; ++r)
          if (row0 + r < M) Cb[(size_t)(row0 + r) * ldc + col] = acc[i][j][r];
      }
    }
  }
}

// ---------------------------------------------------------------------------
// RMSNorm (optionally gated by silu(z)) -> bf16 output
// ---------------------------------------------------------------------------
__global__ __launch_bounds__(256)
void rmsnorm_bf16(const float* __restrict__ x, int xs,
                  const float* __restrict__ gate, int gs,
                  const float* __restrict__ w,
                  unsigned short* __restrict__ out, int len)
{
  __shared__ float red[256];
  const int row = blockIdx.x, tid = threadIdx.x;
  const float* xr = x + (size_t)row * xs;
  const float* gr = gate ? gate + (size_t)row * gs : nullptr;
  float ss = 0.0f;
  for (int i = tid; i < len; i += 256) {
    float v = xr[i];
    if (gr) v *= siluf(gr[i]);
    ss += v * v;
  }
  red[tid] = ss; __syncthreads();
  for (int o = 128; o > 0; o >>= 1) { if (tid < o) red[tid] += red[tid + o]; __syncthreads(); }
  float scale = rsqrtf(red[0] / (float)len + EPSV);
  for (int i = tid; i < len; i += 256) {
    float v = xr[i];
    if (gr) v *= siluf(gr[i]);
    out[(size_t)row * len + i] = f2bf(v * scale * w[i]);
  }
}

// ---------------------------------------------------------------------------
// Causal conv(K=4) + SiLU over xBC slice of proj
// ---------------------------------------------------------------------------
__global__ __launch_bounds__(256)
void conv_silu(const float* __restrict__ proj, const float* __restrict__ cw,
               const float* __restrict__ cb, float* __restrict__ xBCa)
{
  int c = blockIdx.x * 256 + threadIdx.x;   // < 4352
  int t = blockIdx.y;
  float a = cb[c];
#pragma unroll
  for (int i = 0; i < KCONV; ++i) {
    int ts = t - (KCONV - 1) + i;
    if (ts >= 0) a += proj[(size_t)ts * PROJW + DINNER + c] * cw[c * KCONV + i];
  }
  xBCa[(size_t)t * CONVDIM + c] = siluf(a);
}

__global__ __launch_bounds__(64)
void dt_softplus(const float* __restrict__ proj, const float* __restrict__ dtb,
                 float* __restrict__ dt)
{
  int t = blockIdx.x, h = threadIdx.x;
  float v = proj[(size_t)t * PROJW + 2 * DINNER + 2 * NSTATE + h] + dtb[h];
  dt[(size_t)t * NH + h] = (v > 20.0f) ? v : log1pf(__expf(v));
}

__global__ __launch_bounds__(256)
void make_xdt(const float* __restrict__ xBCa, const float* __restrict__ dt,
              float* __restrict__ xdt)
{
  size_t i = (size_t)blockIdx.x * 256 + threadIdx.x;
  int t = (int)(i >> 12), hp = (int)(i & 4095);
  xdt[i] = xBCa[(size_t)t * CONVDIM + hp] * dt[(size_t)t * NH + (hp >> 6)];
}

// cumsum of dt*A per (chunk, head)
__global__ __launch_bounds__(256)
void dacum_scan(const float* __restrict__ dt, const float* __restrict__ A_log,
                float* __restrict__ dac)
{
  __shared__ float s[256];
  int c = blockIdx.x, h = blockIdx.y, tid = threadIdx.x;
  float Ah = -__expf(A_log[h]);
  s[tid] = dt[(size_t)(c * CHUNK + tid) * NH + h] * Ah;
  for (int off = 1; off < 256; off <<= 1) {
    __syncthreads();
    float v = (tid >= off) ? s[tid - off] : 0.0f;
    __syncthreads();
    s[tid] += v;
  }
  dac[(size_t)(c * NH + h) * CHUNK + tid] = s[tid];
}

// ---------------------------------------------------------------------------
// states[c,h][p,n] = sum_l xdt[l,p] * B[l,n] * exp(da_last - da[l])   (WMMA)
// M=64(p) N=128(n) K=256(l); waves 2x4, wave tile 32x32
// ---------------------------------------------------------------------------
__global__ __launch_bounds__(256)
void ssd_states(const float* __restrict__ xdt, const float* __restrict__ xBCa,
                const float* __restrict__ dac, float* __restrict__ states)
{
  __shared__ float decs[256];
  __shared__ __align__(16) unsigned short As[64 * 40];
  __shared__ __align__(16) unsigned short Bs[128 * 40];    // transposed [n][l]
  int c = blockIdx.x, h = blockIdx.y;
  int tid = threadIdx.x, lane = tid & 31, wave = tid >> 5;
  int wm = wave >> 2, wn = wave & 3;

  decs[tid] = dac[(size_t)(c * NH + h) * CHUNK + tid];
  __syncthreads();
  float myda = decs[tid], daL = decs[255];
  __syncthreads();
  decs[tid] = __expf(daL - myda);                          // per-row decay

  f32x8 acc[2][2];
#pragma unroll
  for (int i = 0; i < 2; ++i)
#pragma unroll
    for (int j = 0; j < 2; ++j)
#pragma unroll
      for (int r = 0; r < 8; ++r) acc[i][j][r] = 0.0f;

  for (int k0 = 0; k0 < 256; k0 += 32) {
    __syncthreads();
    // A: 64x32 (xdt transposed); vector read along p, scatter to [p][l]
#pragma unroll
    for (int g = 0; g < 2; ++g) {
      int gi = tid + 256 * g;                 // 512 = 32 l x 16 p4
      int l = gi >> 4, p4 = (gi & 15) * 4;
      f32x4 v = *(const f32x4*)(xdt + (size_t)(c * CHUNK + k0 + l) * DINNER + h * PDIM + p4);
#pragma unroll
      for (int i = 0; i < 4; ++i) As[(p4 + i) * 40 + l] = f2bf(v[i]);
    }
    // B: 32x128 (Bm * decay); vector read along n, scatter to Bt[n][l]
#pragma unroll
    for (int g = 0; g < 4; ++g) {
      int gi = tid + 256 * g;                 // 1024 = 32 l x 32 n4
      int l = gi >> 5, n4 = (gi & 31) * 4;
      float dec = decs[k0 + l];
      f32x4 v = *(const f32x4*)(xBCa + (size_t)(c * CHUNK + k0 + l) * CONVDIM + DINNER + n4);
#pragma unroll
      for (int i = 0; i < 4; ++i) Bs[(n4 + i) * 40 + l] = f2bf(v[i] * dec);
    }
    __syncthreads();
    bf16x16 af[2], bfv[2];
#pragma unroll
    for (int i = 0; i < 2; ++i) af[i] = frag_a(As + (wm * 32 + i * 16) * 40, 40, lane);
#pragma unroll
    for (int j = 0; j < 2; ++j) bfv[j] = frag_bT(Bs + (wn * 32 + j * 16) * 40, lane);
#pragma unroll
    for (int i = 0; i < 2; ++i)
#pragma unroll
      for (int j = 0; j < 2; ++j) acc[i][j] = wmma_bf16(af[i], bfv[j], acc[i][j]);
  }
  size_t base = (size_t)(c * NH + h) * PDIM * NSTATE;
#pragma unroll
  for (int i = 0; i < 2; ++i)
#pragma unroll
    for (int j = 0; j < 2; ++j) {
      int row0 = wm * 32 + i * 16 + ((lane < 16) ? 0 : 8);
      int col  = wn * 32 + j * 16 + (lane & 15);
#pragma unroll
      for (int r = 0; r < 8; ++r)
        states[base + (size_t)(row0 + r) * NSTATE + col] = acc[i][j][r];
    }
}

// sequential inter-chunk scan: prev[c] = carry; carry = carry*exp(daL) + states[c]
__global__ __launch_bounds__(256)
void ssd_scan(const float* __restrict__ states, const float* __restrict__ dac,
              float* __restrict__ prev, int nc)
{
  int h = blockIdx.x, tid = threadIdx.x;
  float carry[32];
#pragma unroll
  for (int j = 0; j < 32; ++j) carry[j] = 0.0f;
  for (int c = 0; c < nc; ++c) {
    float cd = __expf(dac[(size_t)(c * NH + h) * CHUNK + (CHUNK - 1)]);
    size_t base = (size_t)(c * NH + h) * PDIM * NSTATE;
#pragma unroll
    for (int j = 0; j < 32; ++j) {
      size_t e = base + tid + (size_t)j * 256;
      prev[e] = carry[j];
      carry[j] = carry[j] * cd + states[e];
    }
  }
}

// ---------------------------------------------------------------------------
// Fused Y: intra-chunk (CB .* exp(segsum)) @ xdt  +  (C @ prev^T) * out_decay
// + x*D.   M=256(l) N=64(p); 8 waves along M, wave 32x64.
// ---------------------------------------------------------------------------
__global__ __launch_bounds__(256)
void ssd_y(const float* __restrict__ CBb, const float* __restrict__ xdt,
           const float* __restrict__ xBCa, const float* __restrict__ prev,
           const float* __restrict__ dac, const float* __restrict__ Dp,
           float* __restrict__ Y)
{
  __shared__ float da[256];
  __shared__ float eda[256];
  __shared__ __align__(16) unsigned short As[256 * 40];   // 20 KB
  __shared__ __align__(16) unsigned short Bs[64 * 40];    // transposed [p][k]
  int c = blockIdx.x, h = blockIdx.y;
  int tid = threadIdx.x, lane = tid & 31, wave = tid >> 5;

  da[tid] = dac[(size_t)(c * NH + h) * CHUNK + tid];
  __syncthreads();

  f32x8 acc[2][4];
#pragma unroll
  for (int i = 0; i < 2; ++i)
#pragma unroll
    for (int j = 0; j < 4; ++j)
#pragma unroll
      for (int r = 0; r < 8; ++r) acc[i][j][r] = 0.0f;

  // Phase 1: intra-chunk, K = 256 over s
  for (int k0 = 0; k0 < 256; k0 += 32) {
    __syncthreads();
    // A: 256x32 = CB .* L (masked, on the fly); vec4 along s
#pragma unroll
    for (int g = 0; g < 8; ++g) {
      int gi = tid + 256 * g;                 // 2048 = 256 l x 8 s4
      int l = gi >> 3, sl4 = (gi & 7) * 4;
      f32x4 cb = *(const f32x4*)(CBb + (size_t)c * CHUNK * CHUNK + (size_t)l * CHUNK + k0 + sl4);
      float dal = da[l];
      u16x4 hh;
#pragma unroll
      for (int i = 0; i < 4; ++i) {
        int s = k0 + sl4 + i;
        float v = (s <= l) ? cb[i] * __expf(dal - da[s]) : 0.0f;
        hh[i] = f2bf(v);
      }
      *(u16x4*)(&As[l * 40 + sl4]) = hh;
    }
    // B: 32x64 = xdt rows; vec4 along p, scatter to Bt[p][s]
#pragma unroll
    for (int g = 0; g < 2; ++g) {
      int gi = tid + 256 * g;                 // 512 = 32 s x 16 p4
      int sr = gi >> 4, p4 = (gi & 15) * 4;
      f32x4 v = *(const f32x4*)(xdt + (size_t)(c * CHUNK + k0 + sr) * DINNER + h * PDIM + p4);
#pragma unroll
      for (int i = 0; i < 4; ++i) Bs[(p4 + i) * 40 + sr] = f2bf(v[i]);
    }
    __syncthreads();
    bf16x16 af[2], bfv[4];
#pragma unroll
    for (int i = 0; i < 2; ++i) af[i] = frag_a(As + (wave * 32 + i * 16) * 40, 40, lane);
#pragma unroll
    for (int j = 0; j < 4; ++j) bfv[j] = frag_bT(Bs + (j * 16) * 40, lane);
#pragma unroll
    for (int i = 0; i < 2; ++i)
#pragma unroll
      for (int j = 0; j < 4; ++j) acc[i][j] = wmma_bf16(af[i], bfv[j], acc[i][j]);
  }

  eda[tid] = __expf(da[tid]);                 // out_decay, hoisted for phase 2

  // Phase 2: inter-chunk, K = 128 over n: (C[l,n]*exp(da[l])) @ prev[p,n]^T
  for (int k0 = 0; k0 < NSTATE; k0 += 32) {
    __syncthreads();
    // A: 256x32 = C * out_decay; vec4 along n
#pragma unroll
    for (int g = 0; g < 8; ++g) {
      int gi = tid + 256 * g;
      int l = gi >> 3, n4 = (gi & 7) * 4;
      f32x4 v = *(const f32x4*)(xBCa + (size_t)(c * CHUNK + l) * CONVDIM + DINNER + NSTATE + k0 + n4);
      float e = eda[l];
      u16x4 hh;
#pragma unroll
      for (int i = 0; i < 4; ++i) hh[i] = f2bf(v[i] * e);
      *(u16x4*)(&As[l * 40 + n4]) = hh;
    }
    // B: 32x64 from prev[p][n]; vec4 along n (contiguous), direct to Bt[p][n]
#pragma unroll
    for (int g = 0; g < 2; ++g) {
      int gi = tid + 256 * g;                 // 512 = 64 p x 8 n4
      int p = gi >> 3, n4 = (gi & 7) * 4;
      f32x4 v = *(const f32x4*)(prev + (size_t)(c * NH + h) * PDIM * NSTATE + (size_t)p * NSTATE + k0 + n4);
      u16x4 hh;
#pragma unroll
      for (int i = 0; i < 4; ++i) hh[i] = f2bf(v[i]);
      *(u16x4*)(&Bs[p * 40 + n4]) = hh;
    }
    __syncthreads();
    bf16x16 af[2], bfv[4];
#pragma unroll
    for (int i = 0; i < 2; ++i) af[i] = frag_a(As + (wave * 32 + i * 16) * 40, 40, lane);
#pragma unroll
    for (int j = 0; j < 4; ++j) bfv[j] = frag_bT(Bs + (j * 16) * 40, lane);
#pragma unroll
    for (int i = 0; i < 2; ++i)
#pragma unroll
      for (int j = 0; j < 4; ++j) acc[i][j] = wmma_bf16(af[i], bfv[j], acc[i][j]);
  }

  float Dh = Dp[h];
#pragma unroll
  for (int i = 0; i < 2; ++i)
#pragma unroll
    for (int j = 0; j < 4; ++j) {
      int row0 = wave * 32 + i * 16 + ((lane < 16) ? 0 : 8);
      int col  = j * 16 + (lane & 15);
#pragma unroll
      for (int r = 0; r < 8; ++r) {
        int t = c * CHUNK + row0 + r;
        float xv = xBCa[(size_t)t * CONVDIM + h * PDIM + col];
        Y[(size_t)t * DINNER + h * PDIM + col] = acc[i][j][r] + xv * Dh;
      }
    }
}

// ---------------------------------------------------------------------------
// Router: logits = h2 @ W[2048,8]; top-2 softmax -> gates[T,8]
// ---------------------------------------------------------------------------
__global__ __launch_bounds__(32)
void router_top2(const unsigned short* __restrict__ h2bf,
                 const float* __restrict__ rw, float* __restrict__ gates)
{
  int t = blockIdx.x, lane = threadIdx.x;
  float a[NEXP];
#pragma unroll
  for (int e = 0; e < NEXP; ++e) a[e] = 0.0f;
  for (int k = lane; k < DMODEL; k += 32) {
    float hv = bf2f(h2bf[(size_t)t * DMODEL + k]);
#pragma unroll
    for (int e = 0; e < NEXP; ++e) a[e] += hv * rw[(size_t)k * NEXP + e];
  }
#pragma unroll
  for (int e = 0; e < NEXP; ++e)
    for (int off = 16; off > 0; off >>= 1) a[e] += __shfl_down(a[e], off, 32);
  if (lane == 0) {
    int i1 = 0;
#pragma unroll
    for (int e = 1; e < NEXP; ++e) if (a[e] > a[i1]) i1 = e;
    int i2 = (i1 == 0) ? 1 : 0;
#pragma unroll
    for (int e = 0; e < NEXP; ++e) if (e != i1 && a[e] > a[i2]) i2 = e;
    float w2 = __expf(a[i2] - a[i1]);
    float g1 = 1.0f / (1.0f + w2), g2 = 1.0f - g1;
#pragma unroll
    for (int e = 0; e < NEXP; ++e)
      gates[(size_t)t * NEXP + e] = (e == i1) ? g1 : ((e == i2) ? g2 : 0.0f);
  }
}

__global__ __launch_bounds__(256)
void swiglu_bf16(const float* __restrict__ gu, unsigned short* __restrict__ out, int inner)
{
  size_t i = (size_t)blockIdx.x * 256 + threadIdx.x;
  size_t t = i / inner, k = i % inner;
  float a = gu[t * (2 * inner) + k];
  float b = gu[t * (2 * inner) + inner + k];
  out[t * inner + k] = f2bf(siluf(a) * b);
}

__global__ __launch_bounds__(256)
void moe_accum(const float* __restrict__ etmp, const float* __restrict__ gates,
               float* __restrict__ moe, int e, int init)
{
  size_t i = (size_t)blockIdx.x * 256 + threadIdx.x;
  int t = (int)(i >> 11);
  float g = gates[(size_t)t * NEXP + e];
  float v = g * etmp[i];
  moe[i] = init ? v : (moe[i] + v);
}

__global__ __launch_bounds__(256)
void residual_add(const float* __restrict__ a, const float* __restrict__ b,
                  float* __restrict__ out, float mult)
{
  size_t i = (size_t)blockIdx.x * 256 + threadIdx.x;
  out[i] = a[i] + b[i] * mult;
}

__global__ __launch_bounds__(256)
void final_add(const float* __restrict__ resid, const float* __restrict__ moe,
               const float* __restrict__ sh, float* __restrict__ out)
{
  size_t i = (size_t)blockIdx.x * 256 + threadIdx.x;
  out[i] = resid[i] + (moe[i] + sh[i]) * RESM;
}

// ---------------------------------------------------------------------------
extern "C" void kernel_launch(void* const* d_in, const int* in_sizes, int n_in,
                              void* d_out, int out_size, void* d_ws, size_t ws_size,
                              hipStream_t stream)
{
  const float* hs        = (const float*)d_in[1];
  const float* ln1_w     = (const float*)d_in[2];
  const float* in_proj_w = (const float*)d_in[3];
  const float* conv_w    = (const float*)d_in[4];
  const float* conv_b    = (const float*)d_in[5];
  const float* A_log     = (const float*)d_in[6];
  const float* dt_bias   = (const float*)d_in[7];
  const float* D_param   = (const float*)d_in[8];
  const float* mnorm_w   = (const float*)d_in[9];
  const float* outp_w    = (const float*)d_in[10];
  const float* ln2_w     = (const float*)d_in[11];
  const float* router_w  = (const float*)d_in[12];
  const float* moe_gu_w  = (const float*)d_in[13];
  const float* moe_dn_w  = (const float*)d_in[14];
  const float* sh_in_w   = (const float*)d_in[15];
  const float* sh_out_w  = (const float*)d_in[16];

  const int T  = in_sizes[1] / DMODEL;   // 4096
  const int nc = T / CHUNK;              // 16

  float* out_hidden = (float*)d_out;
  float* out_resid  = out_hidden + (size_t)T * DMODEL;

  // ---- workspace layout ----
  char* ws = (char*)d_ws;
  size_t off = 0;
  auto alloc = [&](size_t bytes) { size_t r = off; off = (off + bytes + 255) & ~(size_t)255; return r; };
  unsigned short* hbf   = (unsigned short*)(ws + alloc((size_t)T * DMODEL * 2));
  float*          proj  = (float*)(ws + alloc((size_t)T * PROJW * 4));   // reused as `gu`
  float*          xBCa  = (float*)(ws + alloc((size_t)T * CONVDIM * 4));
  float*          dtb   = (float*)(ws + alloc((size_t)T * NH * 4));
  float*          dac   = (float*)(ws + alloc((size_t)nc * NH * CHUNK * 4));
  float*          xdt   = (float*)(ws + alloc((size_t)T * DINNER * 4));
  float*          CBb   = (float*)(ws + alloc((size_t)nc * CHUNK * CHUNK * 4));
  float*          stbuf = (float*)(ws + alloc((size_t)nc * NH * PDIM * NSTATE * 4));
  float*          prevb = (float*)(ws + alloc((size_t)nc * NH * PDIM * NSTATE * 4));
  float*          Ybuf  = (float*)(ws + alloc((size_t)T * DINNER * 4));
  unsigned short* ybf   = (unsigned short*)(ws + alloc((size_t)T * DINNER * 2));
  float*          mamb  = (float*)(ws + alloc((size_t)T * DMODEL * 4));
  unsigned short* h2bf  = (unsigned short*)(ws + alloc((size_t)T * DMODEL * 2));
  float*          gates = (float*)(ws + alloc((size_t)T * NEXP * 4));
  unsigned short* actbf = (unsigned short*)(ws + alloc((size_t)T * ISH * 2));
  float*          etmp  = (float*)(ws + alloc((size_t)T * DMODEL * 4));
  float*          moeb  = (float*)(ws + alloc((size_t)T * DMODEL * 4));
  float*          shb   = (float*)(ws + alloc((size_t)T * DMODEL * 4));
  // bf16 weight copies
  unsigned short* w_inp = (unsigned short*)(ws + alloc((size_t)DMODEL * PROJW * 2));
  unsigned short* w_out = (unsigned short*)(ws + alloc((size_t)DINNER * DMODEL * 2));
  unsigned short* w_mgu = (unsigned short*)(ws + alloc((size_t)NEXP * DMODEL * 2 * IEXP * 2));
  unsigned short* w_mdn = (unsigned short*)(ws + alloc((size_t)NEXP * IEXP * DMODEL * 2));
  unsigned short* w_shi = (unsigned short*)(ws + alloc((size_t)DMODEL * 2 * ISH * 2));
  unsigned short* w_sho = (unsigned short*)(ws + alloc((size_t)ISH * DMODEL * 2));

  // 0) weight f32 -> bf16 (once per launch, bandwidth-trivial)
  {
    long long n1 = (long long)DMODEL * PROJW;
    long long n2 = (long long)DINNER * DMODEL;
    long long n3 = (long long)NEXP * DMODEL * 2 * IEXP;
    long long n4 = (long long)NEXP * IEXP * DMODEL;
    long long n5 = (long long)DMODEL * 2 * ISH;
    long long n6 = (long long)ISH * DMODEL;
    cvt_bf16<<<(int)(n1 / 2048), 256, 0, stream>>>(in_proj_w, w_inp, n1);
    cvt_bf16<<<(int)(n2 / 2048), 256, 0, stream>>>(outp_w,    w_out, n2);
    cvt_bf16<<<(int)(n3 / 2048), 256, 0, stream>>>(moe_gu_w,  w_mgu, n3);
    cvt_bf16<<<(int)(n4 / 2048), 256, 0, stream>>>(moe_dn_w,  w_mdn, n4);
    cvt_bf16<<<(int)(n5 / 2048), 256, 0, stream>>>(sh_in_w,   w_shi, n5);
    cvt_bf16<<<(int)(n6 / 2048), 256, 0, stream>>>(sh_out_w,  w_sho, n6);
  }

  // 1) ln1 -> bf16 activations
  rmsnorm_bf16<<<T, 256, 0, stream>>>(hs, DMODEL, nullptr, 0, ln1_w, hbf, DMODEL);

  // 2) in_proj: [T,2048] x [2048,8512]
  gemm_wmma<true, true, false><<<dim3((PROJW + 127) / 128, T / 128, 1), 256, 0, stream>>>(
      hbf, w_inp, proj, T, PROJW, DMODEL, DMODEL, PROJW, PROJW, 0, 0, 0);

  // 3) conv + silu over xBC
  conv_silu<<<dim3(CONVDIM / 256, T), 256, 0, stream>>>(proj, conv_w, conv_b, xBCa);

  // 4) dt = softplus(dt_raw + bias);  xdt = x * dt
  dt_softplus<<<T, 64, 0, stream>>>(proj, dt_bias, dtb);
  make_xdt<<<(int)(((size_t)T * DINNER) / 256), 256, 0, stream>>>(xBCa, dtb, xdt);

  // 5) per-(chunk,head) cumsum of dt*A
  dacum_scan<<<dim3(nc, NH), 256, 0, stream>>>(dtb, A_log, dac);

  // 6) CB = C @ B^T per chunk (batched over z)
  gemm_wmma<false, false, true><<<dim3(2, 2, nc), 256, 0, stream>>>(
      xBCa + DINNER + NSTATE, xBCa + DINNER, CBb,
      CHUNK, CHUNK, NSTATE, CONVDIM, CONVDIM, CHUNK,
      (long long)CHUNK * CONVDIM, (long long)CHUNK * CONVDIM, (long long)CHUNK * CHUNK);

  // 7) chunk-local states, 8) inter-chunk scan, 9) fused Y
  ssd_states<<<dim3(nc, NH), 256, 0, stream>>>(xdt, xBCa, dac, stbuf);
  ssd_scan<<<NH, 256, 0, stream>>>(stbuf, dac, prevb, nc);
  ssd_y<<<dim3(nc, NH), 256, 0, stream>>>(CBb, xdt, xBCa, prevb, dac, D_param, Ybuf);

  // 10) gated rmsnorm: rmsnorm(Y * silu(z)) -> bf16
  rmsnorm_bf16<<<T, 256, 0, stream>>>(Ybuf, DINNER, proj /*z*/, PROJW, mnorm_w, ybf, DINNER);

  // 11) out_proj
  gemm_wmma<true, true, false><<<dim3(DMODEL / 128, T / 128, 1), 256, 0, stream>>>(
      ybf, w_out, mamb, T, DMODEL, DINNER, DINNER, DMODEL, DMODEL, 0, 0, 0);

  // 12) residual (-> second output) + ln2 -> bf16
  residual_add<<<(int)(((size_t)T * DMODEL) / 256), 256, 0, stream>>>(hs, mamb, out_resid, RESM);
  rmsnorm_bf16<<<T, 256, 0, stream>>>(out_resid, DMODEL, nullptr, 0, ln2_w, h2bf, DMODEL);

  // 13) router top-2
  router_top2<<<T, 32, 0, stream>>>(h2bf, router_w, gates);

  // 14) MoE (dense, matches reference math)
  for (int e = 0; e < NEXP; ++e) {
    gemm_wmma<true, true, false><<<dim3((2 * IEXP) / 128, T / 128, 1), 256, 0, stream>>>(
        h2bf, w_mgu + (size_t)e * DMODEL * 2 * IEXP, proj,
        T, 2 * IEXP, DMODEL, DMODEL, 2 * IEXP, 2 * IEXP, 0, 0, 0);
    swiglu_bf16<<<(int)(((size_t)T * IEXP) / 256), 256, 0, stream>>>(proj, actbf, IEXP);
    gemm_wmma<true, true, false><<<dim3(DMODEL / 128, T / 128, 1), 256, 0, stream>>>(
        actbf, w_mdn + (size_t)e * IEXP * DMODEL, etmp,
        T, DMODEL, IEXP, IEXP, DMODEL, DMODEL, 0, 0, 0);
    moe_accum<<<(int)(((size_t)T * DMODEL) / 256), 256, 0, stream>>>(etmp, gates, moeb, e, e == 0);
  }

  // 15) shared expert
  gemm_wmma<true, true, false><<<dim3((2 * ISH) / 128, T / 128, 1), 256, 0, stream>>>(
      h2bf, w_shi, proj, T, 2 * ISH, DMODEL, DMODEL, 2 * ISH, 2 * ISH, 0, 0, 0);
  swiglu_bf16<<<(int)(((size_t)T * ISH) / 256), 256, 0, stream>>>(proj, actbf, ISH);
  gemm_wmma<true, true, false><<<dim3(DMODEL / 128, T / 128, 1), 256, 0, stream>>>(
      actbf, w_sho, shb, T, DMODEL, ISH, ISH, DMODEL, DMODEL, 0, 0, 0);

  // 16) final residual -> first output
  final_add<<<(int)(((size_t)T * DMODEL) / 256), 256, 0, stream>>>(out_resid, moeb, shb, out_hidden);
}
